// SimAttn_61237643706653
// MI455X (gfx1250) — matrix operand order repaired and checked by reading, not compiled, not run on hardware
//
#include <hip/hip_runtime.h>

// ---------------------------------------------------------------------------
// Two-pass flash attention with softmax-sharpening for gfx1250 (MI455X).
// f16 WMMA (v_wmma_f32_16x16x32_f16) for both GEMMs, f32 softmax in exp2 domain
// using raw v_exp_f32 (__builtin_amdgcn_exp2f; flush-to-zero tails are exact
// enough for attention weights).
//   pass 1: per-lane partial max/sum-exp2, one cross-lane reduce at the end
//   pass 2: closed-form row max; z2 via ones-WMMA; no reductions at all
// Global->LDS staging is register-pipelined: next tile's loads overlap compute.
// ---------------------------------------------------------------------------

typedef __attribute__((ext_vector_type(16))) _Float16 v16h;
typedef __attribute__((ext_vector_type(8)))  _Float16 v8h;
typedef __attribute__((ext_vector_type(4)))  _Float16 v4h;
typedef __attribute__((ext_vector_type(8)))  float    v8f;
typedef __attribute__((ext_vector_type(4)))  float    v4f;

#define T_SEQ   2048
#define D_MODEL 1024
#define N_HEADS 8
#define DK      128
#define QTILE   128     // q rows per block (8 waves x 16 rows)
#define KTILE   32      // keys per main-loop iteration
#define NWAVES  8
#define QS_STR  136     // LDS row pitch in halves (128 + 8 pad) -> 272B = 17*16
#define KS_STR  136
#define VT_STR  40      // V^T row pitch in halves (32 + 8 pad)  -> 80B  = 5*16
#define PB_STR  40

union AFrag { v16h v; v8h h[2]; };

__device__ __forceinline__ float fexp2(float x) {
  return __builtin_amdgcn_exp2f(x);   // single v_exp_f32, no denorm guard
}

// One-time 16-lane reductions (outside hot loops).
__device__ __forceinline__ float rowmax16(float v) {
  v = fmaxf(v, __shfl_xor(v, 1, 32));
  v = fmaxf(v, __shfl_xor(v, 2, 32));
  v = fmaxf(v, __shfl_xor(v, 4, 32));
  v = fmaxf(v, __shfl_xor(v, 8, 32));
  return v;
}
__device__ __forceinline__ float rowsum16(float v) {
  v += __shfl_xor(v, 1, 32);
  v += __shfl_xor(v, 2, 32);
  v += __shfl_xor(v, 4, 32);
  v += __shfl_xor(v, 8, 32);
  return v;
}

// Contiguous 16-half (32B) LDS fragment read as two aligned b128 loads.
__device__ __forceinline__ v16h ldsB(const _Float16* p) {
  AFrag f;
  f.h[0] = *(const v8h*)(p);
  f.h[1] = *(const v8h*)(p + 8);
  return f.v;
}

__global__ __launch_bounds__(256)
void SimAttn_wmma_kernel(const float* __restrict__ Qg,
                         const float* __restrict__ Kg,
                         const float* __restrict__ Vg,
                         float* __restrict__ Og)
{
  __shared__ _Float16 Qs[QTILE * QS_STR];            // 34816 B
  __shared__ _Float16 Ks[KTILE * KS_STR];            //  8704 B
  __shared__ _Float16 Vt[DK * VT_STR];               // 10240 B (V transposed)
  __shared__ _Float16 Pb[NWAVES * 16 * PB_STR];      // 10240 B (per-wave attn)

  const int tid  = threadIdx.x;
  const int wave = tid >> 5;
  const int lane = tid & 31;
  const int lr   = lane & 15;   // row/col index inside a 16-lane half
  const int g    = lane >> 4;   // which lane half
  const int bh   = blockIdx.y;  // b*N_HEADS + h
  const int qb   = blockIdx.x * QTILE;
  // scale/sqrt(dk) folded with log2(e): scores go straight to exp2 domain.
  const float cs = 0.088388347648318447f * 1.4426950408889634f;

  const size_t base = (size_t)(bh >> 3) * T_SEQ * D_MODEL + (size_t)(bh & 7) * DK;
  const float* Qp = Qg + base;
  const float* Kp = Kg + base;
  const float* Vp = Vg + base;
  float*       Op = Og + base;

  // ---- Stage Q tile (128 x 128 f32 -> f16, row major in LDS) --------------
  #pragma unroll
  for (int i = 0; i < 16; ++i) {
    int idx = i * 256 + tid;                // 0..4095
    int r = idx >> 5, c4 = idx & 31;
    v4f q = *(const v4f*)(Qp + (size_t)(qb + r) * D_MODEL + c4 * 4);
    v4h hq = { (_Float16)q.x, (_Float16)q.y, (_Float16)q.z, (_Float16)q.w };
    *(v4h*)(&Qs[r * QS_STR + c4 * 4]) = hq;
  }
  __syncthreads();

  // Persistent A-fragments of this wave's 16 q rows (4 chunks of K=32).
  AFrag qa[4];
  {
    const _Float16* qrow = &Qs[(wave * 16 + lr) * QS_STR];
    #pragma unroll
    for (int dc = 0; dc < 4; ++dc) {
      qa[dc].h[0] = *(const v8h*)(qrow + dc * 32 + 16 * g);       // K = 16g+j
      qa[dc].h[1] = *(const v8h*)(qrow + dc * 32 + 16 * g + 8);   // K = 16g+8+j
    }
  }

  // ============== PASS 1: per-lane partial rowmax & sum(2^l) ==============
  // l = s*scale*log2e is bounded well below f32 exp2 limits -> no stabilizer.
  float mloc[8], zloc[8];
  #pragma unroll
  for (int r = 0; r < 8; ++r) { mloc[r] = -3.0e38f; zloc[r] = 0.f; }

  v4f kst[4];                                  // pipelined K staging registers
  #pragma unroll
  for (int i = 0; i < 4; ++i) {
    int idx = i * 256 + tid;
    int r = idx >> 5, c4 = idx & 31;
    kst[i] = *(const v4f*)(Kp + (size_t)r * D_MODEL + c4 * 4);
  }

  for (int kt = 0; kt < T_SEQ; kt += KTILE) {
    __syncthreads();
    #pragma unroll
    for (int i = 0; i < 4; ++i) {              // regs -> LDS
      int idx = i * 256 + tid;
      int r = idx >> 5, c4 = idx & 31;
      v4h hk = { (_Float16)kst[i].x, (_Float16)kst[i].y,
                 (_Float16)kst[i].z, (_Float16)kst[i].w };
      *(v4h*)(&Ks[r * KS_STR + c4 * 4]) = hk;
    }
    if (kt + KTILE < T_SEQ) {                  // issue next tile's loads now
      #pragma unroll
      for (int i = 0; i < 4; ++i) {
        int idx = i * 256 + tid;
        int r = idx >> 5, c4 = idx & 31;
        kst[i] = *(const v4f*)(Kp + (size_t)(kt + KTILE + r) * D_MODEL + c4 * 4);
      }
    }
    __syncthreads();

    v8f st0 = {0.f,0.f,0.f,0.f,0.f,0.f,0.f,0.f};
    v8f st1 = {0.f,0.f,0.f,0.f,0.f,0.f,0.f,0.f};
    #pragma unroll
    for (int dc = 0; dc < 4; ++dc) {
      v16h b0 = ldsB(&Ks[lr * KS_STR + dc * 32 + 16 * g]);
      st0 = __builtin_amdgcn_wmma_f32_16x16x32_f16(false, qa[dc].v, false, b0,
                                                   (short)0, st0, false, false);
      v16h b1 = ldsB(&Ks[(16 + lr) * KS_STR + dc * 32 + 16 * g]);
      st1 = __builtin_amdgcn_wmma_f32_16x16x32_f16(false, qa[dc].v, false, b1,
                                                   (short)0, st1, false, false);
    }
    #pragma unroll
    for (int r = 0; r < 8; ++r) {              // lane holds row (r + 8g)
      float l0 = st0[r] * cs;
      float l1 = st1[r] * cs;
      mloc[r] = fmaxf(mloc[r], fmaxf(l0, l1));
      zloc[r] += fexp2(l0) + fexp2(l1);
    }
  }

  // One-time cross-lane reduce; closed-form second-softmax row max:
  //   p_i = 2^l_i / Z1,  max_i(1000*p_i) = 1000*2^ms/Z1 = m2;
  //   e2_i = exp(1000*p_i - m2) = exp2(m2e*2^(l_i-ms) - m2e), m2e = m2*log2e.
  float ms[8], m2e[8];
  #pragma unroll
  for (int r = 0; r < 8; ++r) {
    float msr = rowmax16(mloc[r]);
    float z1r = rowsum16(zloc[r]);
    ms[r]  = msr;
    m2e[r] = 1.4426950408889634f * 1000.f * fexp2(msr) / z1r;
  }

  // ========== PASS 2: streaming sharpened attention, no reductions ========
  v8f o[8], o9;
  const v8f vzero = {0.f,0.f,0.f,0.f,0.f,0.f,0.f,0.f};
  #pragma unroll
  for (int dc = 0; dc < 8; ++dc) o[dc] = vzero;
  o9 = vzero;
  v16h ones;
  #pragma unroll
  for (int i = 0; i < 16; ++i) ones[i] = (_Float16)1.f;

  v4f vst[4];                                  // pipelined K+V staging regs
  #pragma unroll
  for (int i = 0; i < 4; ++i) {
    int idx = i * 256 + tid;
    int r = idx >> 5, c4 = idx & 31;
    kst[i] = *(const v4f*)(Kp + (size_t)r * D_MODEL + c4 * 4);
    vst[i] = *(const v4f*)(Vp + (size_t)r * D_MODEL + c4 * 4);
  }

  for (int kt = 0; kt < T_SEQ; kt += KTILE) {
    __syncthreads();
    #pragma unroll
    for (int i = 0; i < 4; ++i) {              // K regs -> LDS (row major)
      int idx = i * 256 + tid;
      int r = idx >> 5, c4 = idx & 31;
      v4h hk = { (_Float16)kst[i].x, (_Float16)kst[i].y,
                 (_Float16)kst[i].z, (_Float16)kst[i].w };
      *(v4h*)(&Ks[r * KS_STR + c4 * 4]) = hk;
    }
    #pragma unroll
    for (int i = 0; i < 4; ++i) {              // V regs -> LDS (transposed)
      int idx = i * 256 + tid;
      int r = idx >> 5, c4 = idx & 31;         // r = key, c4*4 = d
      int d0 = c4 * 4;
      Vt[(d0 + 0) * VT_STR + r] = (_Float16)vst[i].x;
      Vt[(d0 + 1) * VT_STR + r] = (_Float16)vst[i].y;
      Vt[(d0 + 2) * VT_STR + r] = (_Float16)vst[i].z;
      Vt[(d0 + 3) * VT_STR + r] = (_Float16)vst[i].w;
    }
    if (kt + KTILE < T_SEQ) {                  // issue next tile's loads now
      #pragma unroll
      for (int i = 0; i < 4; ++i) {
        int idx = i * 256 + tid;
        int r = idx >> 5, c4 = idx & 31;
        kst[i] = *(const v4f*)(Kp + (size_t)(kt + KTILE + r) * D_MODEL + c4 * 4);
        vst[i] = *(const v4f*)(Vp + (size_t)(kt + KTILE + r) * D_MODEL + c4 * 4);
      }
    }
    __syncthreads();

    // Recompute both 16-key score tiles.
    v8f st0 = vzero, st1 = vzero;
    #pragma unroll
    for (int dc = 0; dc < 4; ++dc) {
      v16h b0 = ldsB(&Ks[lr * KS_STR + dc * 32 + 16 * g]);
      st0 = __builtin_amdgcn_wmma_f32_16x16x32_f16(false, qa[dc].v, false, b0,
                                                   (short)0, st0, false, false);
      v16h b1 = ldsB(&Ks[(16 + lr) * KS_STR + dc * 32 + 16 * g]);
      st1 = __builtin_amdgcn_wmma_f32_16x16x32_f16(false, qa[dc].v, false, b1,
                                                   (short)0, st1, false, false);
    }

    // Sharpened weights -> per-wave P buffer (C-frag row map: row = r+8g).
    _Float16* pbw = &Pb[wave * 16 * PB_STR];
    #pragma unroll
    for (int r = 0; r < 8; ++r) {
      float ex0 = fexp2(fmaf(st0[r], cs, -ms[r]));
      float ex1 = fexp2(fmaf(st1[r], cs, -ms[r]));
      float e0 = fexp2(fmaf(m2e[r], ex0, -m2e[r]));
      float e1 = fexp2(fmaf(m2e[r], ex1, -m2e[r]));
      _Float16* prow = pbw + (r + 8 * g) * PB_STR;
      prow[lr]      = (_Float16)e0;
      prow[16 + lr] = (_Float16)e1;
    }

    // P (16x32) as A-fragment; V^T chunks + ones column as B-fragments.
    AFrag pa;
    {
      const _Float16* pr = pbw + lr * PB_STR;
      pa.h[0] = *(const v8h*)(pr + 16 * g);
      pa.h[1] = *(const v8h*)(pr + 16 * g + 8);
    }
    #pragma unroll
    for (int dc = 0; dc < 8; ++dc) {
      v16h vf = ldsB(&Vt[(dc * 16 + lr) * VT_STR + 16 * g]);
      o[dc] = __builtin_amdgcn_wmma_f32_16x16x32_f16(false, pa.v, false, vf,
                                                     (short)0, o[dc], false, false);
    }
    o9 = __builtin_amdgcn_wmma_f32_16x16x32_f16(false, pa.v, false, ones,
                                                (short)0, o9, false, false);
  }

  // ---- Normalize by z2 (= o9 row) and store --------------------------------
  #pragma unroll
  for (int r = 0; r < 8; ++r) {
    float inv = 1.f / o9[r];
    float* orow = Op + (size_t)(qb + wave * 16 + r + 8 * g) * D_MODEL;
    #pragma unroll
    for (int dc = 0; dc < 8; ++dc)
      orow[dc * 16 + lr] = o[dc][r] * inv;
  }
}

extern "C" void kernel_launch(void* const* d_in, const int* in_sizes, int n_in,
                              void* d_out, int out_size, void* d_ws, size_t ws_size,
                              hipStream_t stream) {
  (void)in_sizes; (void)n_in; (void)out_size; (void)d_ws; (void)ws_size;
  const float* Q = (const float*)d_in[0];
  const float* K = (const float*)d_in[1];
  const float* V = (const float*)d_in[2];
  float* Out = (float*)d_out;
  dim3 grid(T_SEQ / QTILE, 4 * N_HEADS);   // (16, 32)
  SimAttn_wmma_kernel<<<grid, 256, 0, stream>>>(Q, K, V, Out);
}
